// MultiHeadAttention_53206054863571
// MI455X (gfx1250) — matrix operand order repaired and checked by reading, not compiled, not run on hardware
//
#include <hip/hip_runtime.h>
#include <hip/hip_bf16.h>

// Problem constants (match reference)
#define B_   4
#define S_   2048
#define D_   1024
#define H_   16
#define DK_  64

#ifndef __has_builtin
#define __has_builtin(x) 0
#endif
#if __has_builtin(__builtin_amdgcn_global_load_async_to_lds_b128) && \
    __has_builtin(__builtin_amdgcn_s_wait_asynccnt)
#define USE_ASYNC_LDS 1
#else
#define USE_ASYNC_LDS 0
#endif

typedef __attribute__((ext_vector_type(16))) _Float16 v16h;
typedef __attribute__((ext_vector_type(8)))  _Float16 v8h;
typedef __attribute__((ext_vector_type(4)))  _Float16 v4h;
typedef __attribute__((ext_vector_type(8)))  float    v8f;
typedef __attribute__((ext_vector_type(4)))  float    v4f;
typedef __attribute__((ext_vector_type(4)))  int      i4;

#if USE_ASYNC_LDS
// Pointer types matching the async-copy builtin's parameters:
//   arg0: i4 in address_space(1) (global), arg1: i4 in address_space(3) (LDS)
typedef i4 __attribute__((address_space(1)))* gi4p;
typedef i4 __attribute__((address_space(3)))* li4p;
typedef const void __attribute__((address_space(1)))* gvoidp;
typedef void __attribute__((address_space(3)))* lvoidp;
#endif

static __device__ __forceinline__ v16h cat16(v8h lo, v8h hi) {
  v16h r;
#pragma unroll
  for (int i = 0; i < 8; ++i) { r[i] = lo[i]; r[i + 8] = hi[i]; }
  return r;
}

static __device__ __forceinline__ v8f wmma_f16(v16h a, v16h b, v8f c) {
  // D = A(16x32 f16) * B(32x16 f16) + C(16x16 f32)
  return __builtin_amdgcn_wmma_f32_16x16x32_f16(
      /*neg_a=*/false, a, /*neg_b=*/false, b,
      /*c_mod=*/(short)0, c, /*reuse_a=*/false, /*reuse_b=*/false);
}

// ---------------------------------------------------------------------------
// fp32 -> f16 conversion, 4 elements per thread
// ---------------------------------------------------------------------------
__global__ __launch_bounds__(256) void cvt_f32_f16_x4(
    const float* __restrict__ in, _Float16* __restrict__ out, int n4) {
  int i = blockIdx.x * blockDim.x + threadIdx.x;
  if (i < n4) {
    v4f v = *(const v4f*)(in + (size_t)i * 4);
    v4h h;
#pragma unroll
    for (int k = 0; k < 4; ++k) h[k] = (_Float16)v[k];
    *(v4h*)(out + (size_t)i * 4) = h;
  }
}

// ---------------------------------------------------------------------------
// WMMA GEMM: Out = X[8192,1024] @ W^T + bias, W stored [N=1024, K=1024] row-major.
// Block = 128 threads = 4 waves; block tile 128(M) x 64(N); wave tile 32(M) x 64(N).
// B tile (64x32 f16, 4KB) is block-shared: staged into LDS with gfx1250 async
// copies (double-buffered) when available, else loaded directly per wave.
// MODE 0: f16 out, layout [B,H,S,Dk]   (Q, K)
// MODE 1: f16 out, layout [B,H,Dk,S]   (V transposed for the PV WMMA)
// MODE 2: f32 out, row-major [B*S, D]  (final projection into d_out)
// ---------------------------------------------------------------------------
template <int MODE>
__global__ __launch_bounds__(128) void gemm16(
    const _Float16* __restrict__ X, const _Float16* __restrict__ W,
    const float* __restrict__ bias, void* __restrict__ OutV) {
#if USE_ASYNC_LDS
  __shared__ _Float16 ldsB[2][64 * 32];  // [buf][n-row][k], 4KB per buffer
#endif
  const int tid  = threadIdx.x;
  const int lane = tid & 31;
  const int wave = tid >> 5;
  const int lh = lane & 15;   // row/col-in-tile selector
  const int hi = lane >> 4;   // half-wave selector
  const int mbase = blockIdx.y * 128 + wave * 32;
  const int nbase = blockIdx.x * 64;

  // A fragments: row M = mbase+{0,16}+lh; lanes 0-15 hold k=+0..7,+16..23,
  // lanes 16-31 hold k=+8..15,+24..31.
  const _Float16* arow0 = X + (size_t)(mbase + lh) * D_ + hi * 8;
  const _Float16* arow1 = arow0 + (size_t)16 * D_;

  v8f c[2][4];
#pragma unroll
  for (int i = 0; i < 2; ++i)
#pragma unroll
    for (int j = 0; j < 4; ++j)
      c[i][j] = (v8f){0.f, 0.f, 0.f, 0.f, 0.f, 0.f, 0.f, 0.f};

#if USE_ASYNC_LDS
  // 4KB B tile = 256 16-byte chunks; each of 128 threads copies chunks tid, tid+128.
  // chunk ch: n-row = ch>>2, k-offset = (ch&3)*8 halves.
  auto issue_copy = [&](int k0, int buf) {
#pragma unroll
    for (int u = 0; u < 2; ++u) {
      const int ch = tid + u * 128;
      const int row = ch >> 2, koff = (ch & 3) * 8;
      const _Float16* src = W + (size_t)(nbase + row) * D_ + k0 + koff;
      _Float16* dst = &ldsB[buf][row * 32 + koff];
      __builtin_amdgcn_global_load_async_to_lds_b128(
          (gi4p)(gvoidp)src, (li4p)(lvoidp)dst, 0, 0);
    }
  };
  issue_copy(0, 0);
  for (int it = 0, k0 = 0; k0 < D_; ++it, k0 += 32) {
    const int cur = it & 1;
    if (k0 + 32 < D_) {
      issue_copy(k0 + 32, cur ^ 1);
      __builtin_amdgcn_s_wait_asynccnt(2);  // in-order: oldest batch (cur) done
    } else {
      __builtin_amdgcn_s_wait_asynccnt(0);
    }
    __syncthreads();  // all waves' copies into cur are resident

    __builtin_prefetch(arow0 + k0 + 128, 0, 3);
    __builtin_prefetch(arow1 + k0 + 128, 0, 3);
    v16h a0 = cat16(*(const v8h*)(arow0 + k0), *(const v8h*)(arow0 + k0 + 16));
    v16h a1 = cat16(*(const v8h*)(arow1 + k0), *(const v8h*)(arow1 + k0 + 16));
#pragma unroll
    for (int j = 0; j < 4; ++j) {
      const _Float16* bp = &ldsB[cur][(j * 16 + lh) * 32 + hi * 8];
      v16h b = cat16(*(const v8h*)bp, *(const v8h*)(bp + 16));
      c[0][j] = wmma_f16(a0, b, c[0][j]);
      c[1][j] = wmma_f16(a1, b, c[1][j]);
    }
    __syncthreads();  // everyone done reading cur before it is overwritten
  }
#else
  for (int k0 = 0; k0 < D_; k0 += 32) {
    __builtin_prefetch(arow0 + k0 + 128, 0, 3);
    __builtin_prefetch(arow1 + k0 + 128, 0, 3);
    v16h a0 = cat16(*(const v8h*)(arow0 + k0), *(const v8h*)(arow0 + k0 + 16));
    v16h a1 = cat16(*(const v8h*)(arow1 + k0), *(const v8h*)(arow1 + k0 + 16));
#pragma unroll
    for (int j = 0; j < 4; ++j) {
      // B fragment: col N = nbase+16j+lh; lanes 0-15 hold k=0..7,16..23, 16-31 k=8..15,24..31.
      const _Float16* brow = W + (size_t)(nbase + j * 16 + lh) * D_ + k0 + hi * 8;
      v16h b = cat16(*(const v8h*)brow, *(const v8h*)(brow + 16));
      c[0][j] = wmma_f16(a0, b, c[0][j]);
      c[1][j] = wmma_f16(a1, b, c[1][j]);
    }
  }
#endif

#pragma unroll
  for (int j = 0; j < 4; ++j) {
    const int n = nbase + j * 16 + lh;
    const float bn = bias[n];
#pragma unroll
    for (int i = 0; i < 2; ++i) {
#pragma unroll
      for (int r = 0; r < 8; ++r) {
        const int m = mbase + i * 16 + r + hi * 8;  // C/D: VGPR r -> rows r / r+8
        const float v = c[i][j][r] + bn;
        if (MODE == 2) {
          ((float*)OutV)[(size_t)m * D_ + n] = v;
        } else {
          const int b = m >> 11, s = m & (S_ - 1);
          const int h = n >> 6, d = n & (DK_ - 1);
          _Float16* O = (_Float16*)OutV;
          if (MODE == 0)
            O[(((size_t)(b * H_ + h)) * S_ + s) * DK_ + d] = (_Float16)v;
          else
            O[(((size_t)(b * H_ + h)) * DK_ + d) * S_ + s] = (_Float16)v;
        }
      }
    }
  }
}

// ---------------------------------------------------------------------------
// Flash attention, causal. One wave owns a 16-row Q block for one (b,h).
// Keys processed 32 at a time (two 16x16 score tiles -> one 16x32 P A-fragment).
// grid = B*H*(S/16)/4 = 2048 blocks, 128 threads (4 waves).
// ---------------------------------------------------------------------------
__global__ __launch_bounds__(128) void flash16(
    const _Float16* __restrict__ Q, const _Float16* __restrict__ K,
    const _Float16* __restrict__ Vt, _Float16* __restrict__ Ctx) {
  __shared__ _Float16 plds[4][16 * 32];  // per-wave P staging (C-layout -> A-layout)

  const int lane = threadIdx.x & 31;
  const int wave = threadIdx.x >> 5;
  const int lh = lane & 15;
  const int hi = lane >> 4;
  const int wid = blockIdx.x * 4 + wave;
  const int qb = wid & (S_ / 16 - 1);  // 0..127
  const int bh = wid >> 7;             // 0..63
  const int sbase = qb * 16;

  // Q A-fragments: d=0..31 and d=32..63
  const _Float16* qrow = Q + ((size_t)bh * S_ + sbase + lh) * DK_ + hi * 16;
  const v16h aq0 = cat16(*(const v8h*)(qrow), *(const v8h*)(qrow + 8));
  const v16h aq1 = cat16(*(const v8h*)(qrow + 32), *(const v8h*)(qrow + 40));

  float mrun[8], lrun[8];
#pragma unroll
  for (int r = 0; r < 8; ++r) { mrun[r] = -3.0e38f; lrun[r] = 0.f; }
  v8f acc[4];
#pragma unroll
  for (int j = 0; j < 4; ++j) acc[j] = (v8f){0.f, 0.f, 0.f, 0.f, 0.f, 0.f, 0.f, 0.f};

  _Float16* pl = &plds[wave][0];
  const float scale = 0.125f;  // 1/sqrt(Dk)
  const int rowb = sbase + hi * 8;

  for (int t0 = 0; t0 < sbase + 16; t0 += 32) {
    // ---- scores: S = Q @ K^T (two 16x16 tiles: keys t0..+15 and t0+16..+31)
    const _Float16* kb0 = K + ((size_t)bh * S_ + t0 + lh) * DK_ + hi * 16;
    const _Float16* kb1 = kb0 + 16 * DK_;
    v16h b0lo = cat16(*(const v8h*)(kb0), *(const v8h*)(kb0 + 8));
    v16h b0hi = cat16(*(const v8h*)(kb0 + 32), *(const v8h*)(kb0 + 40));
    v16h b1lo = cat16(*(const v8h*)(kb1), *(const v8h*)(kb1 + 8));
    v16h b1hi = cat16(*(const v8h*)(kb1 + 32), *(const v8h*)(kb1 + 40));
    v8f z = (v8f){0.f, 0.f, 0.f, 0.f, 0.f, 0.f, 0.f, 0.f};
    v8f s0 = wmma_f16(aq0, b0lo, z);
    s0 = wmma_f16(aq1, b0hi, s0);
    v8f s1 = wmma_f16(aq0, b1lo, z);
    s1 = wmma_f16(aq1, b1hi, s1);

    // ---- causal mask + scale, per-row running max
    const int c0 = t0 + lh, c1 = t0 + 16 + lh;
    float mx[8];
#pragma unroll
    for (int r = 0; r < 8; ++r) {
      const int row = rowb + r;
      float v0 = (c0 <= row) ? s0[r] * scale : -3.0e38f;
      float v1 = (c1 <= row) ? s1[r] * scale : -3.0e38f;
      s0[r] = v0; s1[r] = v1;
      mx[r] = fmaxf(v0, v1);
    }
#pragma unroll
    for (int r = 0; r < 8; ++r) {
#pragma unroll
      for (int off = 8; off > 0; off >>= 1)
        mx[r] = fmaxf(mx[r], __shfl_xor(mx[r], off, 32));  // stays within 16-lane half
    }

    // ---- online softmax update
    float alpha[8], rs[8];
#pragma unroll
    for (int r = 0; r < 8; ++r) {
      const float mn = fmaxf(mrun[r], mx[r]);
      alpha[r] = __expf(mrun[r] - mn);
      mrun[r] = mn;
      s0[r] = __expf(s0[r] - mn);
      s1[r] = __expf(s1[r] - mn);
      rs[r] = s0[r] + s1[r];
    }
#pragma unroll
    for (int r = 0; r < 8; ++r) {
#pragma unroll
      for (int off = 8; off > 0; off >>= 1)
        rs[r] += __shfl_xor(rs[r], off, 32);
      lrun[r] = lrun[r] * alpha[r] + rs[r];
    }
#pragma unroll
    for (int j = 0; j < 4; ++j)
#pragma unroll
      for (int r = 0; r < 8; ++r) acc[j][r] *= alpha[r];

    // ---- P (C-layout f32) -> LDS (row-major 16x32 f16) -> A-fragment.
    // Same-wave DS ops are in-order; per-wave private tile, no barrier needed.
#pragma unroll
    for (int r = 0; r < 8; ++r) {
      pl[(r + hi * 8) * 32 + lh] = (_Float16)s0[r];
      pl[(r + hi * 8) * 32 + lh + 16] = (_Float16)s1[r];
    }
    v16h pa = cat16(*(const v8h*)(pl + lh * 32 + hi * 16),
                    *(const v8h*)(pl + lh * 32 + hi * 16 + 8));

    // ---- ctx += P @ V  (V stored transposed [bh][d][t] -> contiguous key loads)
#pragma unroll
    for (int j = 0; j < 4; ++j) {
      const _Float16* vrow = Vt + ((size_t)bh * DK_ + j * 16 + lh) * S_ + t0 + hi * 16;
      v16h bv = cat16(*(const v8h*)vrow, *(const v8h*)(vrow + 8));
      acc[j] = wmma_f16(pa, bv, acc[j]);
    }
  }

  // ---- normalize and store ctx as f16 [B*S, D] row-major
  const int b = bh >> 4, h = bh & 15;
#pragma unroll
  for (int r = 0; r < 8; ++r) {
    const float inv = 1.0f / lrun[r];
    const size_t row = (size_t)(b * S_ + sbase + r + hi * 8);
#pragma unroll
    for (int j = 0; j < 4; ++j)
      Ctx[row * D_ + h * DK_ + j * 16 + lh] = (_Float16)(acc[j][r] * inv);
  }
}

// ---------------------------------------------------------------------------
// Host-side orchestration
// ---------------------------------------------------------------------------
extern "C" void kernel_launch(void* const* d_in, const int* in_sizes, int n_in,
                              void* d_out, int out_size, void* d_ws, size_t ws_size,
                              hipStream_t stream) {
  (void)in_sizes; (void)n_in; (void)out_size; (void)ws_size;
  const float* x  = (const float*)d_in[0];
  const float* wq = (const float*)d_in[1];
  const float* bq = (const float*)d_in[2];
  const float* wk = (const float*)d_in[3];
  const float* bk = (const float*)d_in[4];
  const float* wv = (const float*)d_in[5];
  const float* bv = (const float*)d_in[6];
  const float* wo = (const float*)d_in[7];
  const float* bo = (const float*)d_in[8];
  float* out = (float*)d_out;

  const size_t NX = (size_t)B_ * S_ * D_;        // 8,388,608
  const size_t NW = (size_t)D_ * D_;             // 1,048,576
  const size_t NQ = (size_t)B_ * H_ * S_ * DK_;  // 8,388,608

  _Float16* base = (_Float16*)d_ws;
  _Float16* xh   = base;                 // [B*S, D]
  _Float16* wqh  = xh + NX;              // [D, D]
  _Float16* wkh  = wqh + NW;
  _Float16* wvh  = wkh + NW;
  _Float16* woh  = wvh + NW;
  _Float16* Qh   = woh + NW;             // [B,H,S,Dk]
  _Float16* Kh   = Qh + NQ;              // [B,H,S,Dk]
  _Float16* Vt   = Kh + NQ;              // [B,H,Dk,S]
  _Float16* ctxh = Vt + NQ;              // [B*S, D]

  // 1) fp32 -> f16 conversions
  cvt_f32_f16_x4<<<(int)(NX / 4 / 256), 256, 0, stream>>>(x, xh, (int)(NX / 4));
  cvt_f32_f16_x4<<<(int)(NW / 4 / 256), 256, 0, stream>>>(wq, wqh, (int)(NW / 4));
  cvt_f32_f16_x4<<<(int)(NW / 4 / 256), 256, 0, stream>>>(wk, wkh, (int)(NW / 4));
  cvt_f32_f16_x4<<<(int)(NW / 4 / 256), 256, 0, stream>>>(wv, wvh, (int)(NW / 4));
  cvt_f32_f16_x4<<<(int)(NW / 4 / 256), 256, 0, stream>>>(wo, woh, (int)(NW / 4));

  // 2) Q/K/V projections (WMMA GEMMs), block tile 128x64
  dim3 gg(D_ / 64, (B_ * S_) / 128);  // (16, 64)
  gemm16<0><<<gg, 128, 0, stream>>>(xh, wqh, bq, (void*)Qh);
  gemm16<0><<<gg, 128, 0, stream>>>(xh, wkh, bk, (void*)Kh);
  gemm16<1><<<gg, 128, 0, stream>>>(xh, wvh, bv, (void*)Vt);

  // 3) causal flash attention
  flash16<<<(B_ * H_ * (S_ / 16)) / 4, 128, 0, stream>>>(Qh, Kh, Vt, ctxh);

  // 4) output projection -> fp32 d_out
  gemm16<2><<<gg, 128, 0, stream>>>(ctxh, woh, bo, (void*)out);
}
